// Attention_Layer_29154238005555
// MI455X (gfx1250) — compile-verified, matrix operand order checked by reading, not run on hardware
//
#include <hip/hip_runtime.h>
#include <hip/hip_bf16.h>

// ---------------------------------------------------------------------------
// Attention layer for MI455X (gfx1250), bf16 WMMA + async-LDS pipeline.
//   B=8, S=2048, H=1024 (single head, head_dim = 1024)
// Workspace layout (bytes), total ~202 MB:
//   X16   @ 0           : B*S*H bf16      (33,554,432)
//   W16   @ 33554432    : 3*H*H bf16      ( 6,291,456)  [Wq|Wk|Wv]
//   Q16   @ 39845888    : B*S*H bf16      (33,554,432)
//   K16   @ 73400320    : B*S*H bf16      (33,554,432)
//   Vt16  @ 106954752   : B*H*S bf16      (33,554,432)  (V transposed [b][d][s])
//   Mtile @ 140509184   : B*S*(S/64) f32  ( 2,097,152)
//   Ltile @ 142606336   : B*S*(S/64) f32  ( 2,097,152)
//   P16   @ 144703488   : B*S*S bf16      (67,108,864)
// ---------------------------------------------------------------------------

#define BB 8
#define SS 2048
#define HH 1024
#define NEGV   (-1e9f)
#define SCALE  (0.03125f)      // 1/sqrt(1024)
#define NKT    (SS / 64)       // 32 k-chunks of 64 cols per row
#define LDB    40              // LDS tile row stride in bf16 elems (80 B, padded)

typedef __attribute__((ext_vector_type(16))) __bf16 v16bf;
typedef __attribute__((ext_vector_type(8)))  __bf16 v8bf;
typedef __attribute__((ext_vector_type(4)))  __bf16 v4bf;
typedef __attribute__((ext_vector_type(8)))  float  v8f;
typedef int v4i_vs __attribute__((vector_size(16)));   // matches builtin param

union Frag16 { v16bf v; v8bf h[2]; };

// --- fragment loaders (ISA 7.12.2 16-bit layouts) ---------------------------
// lanes 0-15 hold K = kbase+0..7 & +16..23 ; lanes 16-31 hold +8..15 & +24..31
__device__ __forceinline__ v16bf ldfrag(const __bf16* __restrict__ base, int ld,
                                        int row, int kbase, int half) {
  const __bf16* p = base + (size_t)row * ld + (size_t)kbase + half * 8;
  Frag16 f;
  f.h[0] = *(const v8bf*)(p);
  f.h[1] = *(const v8bf*)(p + 16);
  return f.v;
}

// fragment from a staged 64-row x 32-col LDS tile (row stride LDB elems)
__device__ __forceinline__ v16bf ldfrag_lds(const __bf16* base, int row, int half) {
  const __bf16* p = base + row * LDB + half * 8;
  Frag16 f;
  f.h[0] = *(const v8bf*)(p);
  f.h[1] = *(const v8bf*)(p + 16);
  return f.v;
}

__device__ __forceinline__ v8f wmma_bf16(v16bf a, v16bf b, v8f c) {
  return __builtin_amdgcn_wmma_f32_16x16x32_bf16(false, a, false, b,
                                                 (short)0, c, false, false);
}

// --- CDNA5 async global->LDS (ASYNCcnt) -------------------------------------
// Builtin takes generic v4i* pointers; keep pointer provenance intact so the
// backend can infer global (arg0) and LDS (arg1) address spaces.
__device__ __forceinline__ void async_ld16(const void* g, void* l) {
#if __has_builtin(__builtin_amdgcn_global_load_async_to_lds_b128)
  __builtin_amdgcn_global_load_async_to_lds_b128((v4i_vs*)g, (v4i_vs*)l, 0, 0);
#else
  unsigned lo = (unsigned)(unsigned long long)l;   // low 32 bits = LDS offset
  asm volatile("global_load_async_to_lds_b128 %0, %1, off"
               :: "v"(lo), "v"(g) : "memory");
#endif
}

__device__ __forceinline__ void wait_async0() {
#if __has_builtin(__builtin_amdgcn_s_wait_asynccnt)
  __builtin_amdgcn_s_wait_asynccnt(0);
#else
  asm volatile("s_wait_asynccnt 0" ::: "memory");
#endif
}

// Stage a 64x32 bf16 tile (row stride ld) into LDS, one b128 per thread.
__device__ __forceinline__ void stage_tile(const __bf16* __restrict__ src, int ld,
                                           int row0, int kbase, __bf16* lds, int tid) {
  int r   = tid >> 2;     // 0..63
  int seg = tid & 3;      // 4 x 8 elems
  const __bf16* p = src + (size_t)(row0 + r) * ld + kbase + seg * 8;
  async_ld16(p, lds + r * LDB + seg * 8);
}

// ---------------------------------------------------------------------------
// Kernel 1: fp32 -> bf16 conversion of X, Wq, Wk, Wv
// ---------------------------------------------------------------------------
__global__ __launch_bounds__(256) void cvt_kernel(
    const float* __restrict__ X, const float* __restrict__ Wq,
    const float* __restrict__ Wk, const float* __restrict__ Wv,
    __bf16* __restrict__ X16, __bf16* __restrict__ W16) {
  const size_t X4 = (size_t)BB * SS * HH / 4;
  const size_t W4 = (size_t)HH * HH / 4;
  size_t i = (size_t)blockIdx.x * blockDim.x + threadIdx.x;
  if (i >= X4 + 3 * W4) return;
  const float* src;
  __bf16* dst;
  size_t off;
  if (i < X4)               { src = X;  dst = X16;               off = i; }
  else if (i < X4 + W4)     { src = Wq; dst = W16;               off = i - X4; }
  else if (i < X4 + 2 * W4) { src = Wk; dst = W16 + HH * HH;     off = i - X4 - W4; }
  else                      { src = Wv; dst = W16 + 2 * HH * HH; off = i - X4 - 2 * W4; }
  float4 f = ((const float4*)src)[off];
  v4bf o = { (__bf16)f.x, (__bf16)f.y, (__bf16)f.z, (__bf16)f.w };
  *(v4bf*)(dst + 4 * off) = o;
}

// ---------------------------------------------------------------------------
// Kernel 2: QKV projection.  Block = 8 waves: 256 rows x 64 cols of one matrix.
//   B (weight) tile async-staged to LDS, double buffered; wave = 32x64 strip.
// ---------------------------------------------------------------------------
__global__ __launch_bounds__(256) void qkv_kernel(
    const __bf16* __restrict__ X16, const __bf16* __restrict__ W16,
    __bf16* __restrict__ Q16, __bf16* __restrict__ K16,
    __bf16* __restrict__ Vt16) {
  __shared__ __bf16 ldsB[2][64 * LDB];
  const int tid  = threadIdx.x;
  const int lane = tid & 31;
  const int wave = tid >> 5;
  const int half = lane >> 4;
  const int n    = lane & 15;
  const int mb   = half * 8;
  // blocks: mat(3) x rowBlk(64, 256 rows each) x colStrip(16, 64 cols each)
  int blk = blockIdx.x;
  int mat = blk / (64 * 16);
  int rem = blk % (64 * 16);
  int rb  = rem / 16;
  int ct  = rem % 16;
  int d0  = ct * 64;
  int wrow = rb * 256 + wave * 32;           // this wave's first row
  const __bf16* W = W16 + (size_t)mat * HH * HH;

  v8f acc[2][4] = {};
  stage_tile(W, HH, d0, 0, ldsB[0], tid);
  for (int it = 0; it < HH / 32; ++it) {
    int h = it * 32;
    wait_async0();
    __syncthreads();                          // tile 'it' visible everywhere
    if (it + 1 < HH / 32)
      stage_tile(W, HH, d0, h + 32, ldsB[(it + 1) & 1], tid);
    const __bf16* Bt = ldsB[it & 1];
    v16bf a0 = ldfrag(X16, HH, wrow + n, h, half);
    v16bf a1 = ldfrag(X16, HH, wrow + 16 + n, h, half);
#pragma unroll
    for (int j = 0; j < 4; ++j) {
      v16bf bf = ldfrag_lds(Bt, 16 * j + n, half);
      acc[0][j] = wmma_bf16(a0, bf, acc[0][j]);
      acc[1][j] = wmma_bf16(a1, bf, acc[1][j]);
    }
  }

  if (mat < 2) {
    __bf16* dst = (mat == 0) ? Q16 : K16;
#pragma unroll
    for (int i = 0; i < 2; ++i)
#pragma unroll
      for (int r = 0; r < 8; ++r) {
        size_t base = (size_t)(wrow + 16 * i + mb + r) * HH + d0 + n;
#pragma unroll
        for (int j = 0; j < 4; ++j) dst[base + 16 * j] = (__bf16)acc[i][j][r];
      }
  } else {
    int bidx = wrow / SS;
    int s0   = wrow % SS;
    __bf16* dstb = Vt16 + (size_t)bidx * HH * SS;
#pragma unroll
    for (int i = 0; i < 2; ++i)
#pragma unroll
      for (int r = 0; r < 8; ++r) {
        int s = s0 + 16 * i + mb + r;
#pragma unroll
        for (int j = 0; j < 4; ++j)
          dstb[(size_t)(d0 + 16 * j + n) * SS + s] = (__bf16)acc[i][j][r];
      }
  }
}

// ---------------------------------------------------------------------------
// Kernel 3: scores + tile-local softmax.  Wave = 32 q-rows x 64 k-cols.
// ---------------------------------------------------------------------------
__global__ __launch_bounds__(256) void score_kernel(
    const __bf16* __restrict__ Q16, const __bf16* __restrict__ K16,
    const unsigned char* __restrict__ mask,
    __bf16* __restrict__ P16, float* __restrict__ Mtile,
    float* __restrict__ Ltile) {
  const int lane = threadIdx.x & 31;
  const int half = lane >> 4;
  const int n    = lane & 15;
  const int mb   = half * 8;
  int job = blockIdx.x * (blockDim.x >> 5) + (threadIdx.x >> 5);
  // jobs: b (8) x q-tile (64, 32 rows) x k-chunk (32, 64 cols)
  int b   = job / (64 * NKT);
  int rem = job % (64 * NKT);
  int qt  = rem / NKT;
  int kt  = rem % NKT;
  int q0  = qt * 32;
  int k0  = kt * 64;
  const __bf16* Qb = Q16 + (size_t)b * SS * HH;
  const __bf16* Kb = K16 + (size_t)b * SS * HH;

  v8f c[2][4] = {};
  for (int d = 0; d < HH; d += 32) {
    v16bf a0 = ldfrag(Qb, HH, q0 + n, d, half);
    v16bf a1 = ldfrag(Qb, HH, q0 + 16 + n, d, half);
    if (d + 32 < HH) {   // prefetch next d-step fragments (global_prefetch_b8)
      __builtin_prefetch(Qb + (size_t)(q0 + n) * HH + d + 32, 0, 0);
      __builtin_prefetch(Kb + (size_t)(k0 + n) * HH + d + 32, 0, 0);
    }
#pragma unroll
    for (int j = 0; j < 4; ++j) {
      v16bf bf = ldfrag(Kb, HH, k0 + 16 * j + n, d, half);
      c[0][j] = wmma_bf16(a0, bf, c[0][j]);
      c[1][j] = wmma_bf16(a1, bf, c[1][j]);
    }
  }

#pragma unroll
  for (int i = 0; i < 2; ++i)
#pragma unroll
    for (int r = 0; r < 8; ++r) {
      int q = q0 + 16 * i + mb + r;
      const unsigned char* mrow = mask + ((size_t)b * SS + q) * SS + k0;
      float s[4];
#pragma unroll
      for (int j = 0; j < 4; ++j)
        s[j] = mrow[16 * j + n] ? NEGV : c[i][j][r] * SCALE;
      // row max over the 64 k-cols of this chunk
      float m = fmaxf(fmaxf(s[0], s[1]), fmaxf(s[2], s[3]));
#pragma unroll
      for (int o = 8; o >= 1; o >>= 1) m = fmaxf(m, __shfl_xor(m, o, 32));
      float p[4], l = 0.f;
#pragma unroll
      for (int j = 0; j < 4; ++j) { p[j] = __expf(s[j] - m); l += p[j]; }
#pragma unroll
      for (int o = 8; o >= 1; o >>= 1) l += __shfl_xor(l, o, 32);
      __bf16* Prow = P16 + ((size_t)b * SS + q) * SS + k0;
#pragma unroll
      for (int j = 0; j < 4; ++j) Prow[16 * j + n] = (__bf16)p[j];
      if (n == 0) {   // lanes 0 and 16 cover distinct rows
        size_t st = ((size_t)b * SS + q) * NKT + kt;
        Mtile[st] = m;
        Ltile[st] = l;
      }
    }
}

// ---------------------------------------------------------------------------
// Kernel 4: per-row stat reduction + flash-style in-place rescale of P.
// ---------------------------------------------------------------------------
__global__ __launch_bounds__(256) void finalize_kernel(
    const float* __restrict__ Mtile, const float* __restrict__ Ltile,
    __bf16* __restrict__ P16) {
  int row = blockIdx.x;                  // b*S + q
  __shared__ float sm[NKT];
  __shared__ float sl[NKT];
  __shared__ float fm, fl;
  int t = threadIdx.x;
  if (t < NKT) {
    sm[t] = Mtile[(size_t)row * NKT + t];
    sl[t] = Ltile[(size_t)row * NKT + t];
  }
  __syncthreads();
  if (t == 0) {
    float m = -__builtin_inff();
    for (int i = 0; i < NKT; ++i) m = fmaxf(m, sm[i]);
    float l = 0.f;
    for (int i = 0; i < NKT; ++i) l += sl[i] * __expf(sm[i] - m);
    fm = m;
    fl = l;
  }
  __syncthreads();
  float m   = fm;
  float inv = 1.0f / fl;
  __bf16* Prow = P16 + (size_t)row * SS;
  for (int k = t; k < SS; k += blockDim.x) {
    float sc = __expf(sm[k >> 6] - m) * inv;
    Prow[k] = (__bf16)((float)Prow[k] * sc);
  }
}

// ---------------------------------------------------------------------------
// Kernel 5: O = P @ V.  Block = 8 waves: 256 q-rows x 64 d-cols, Vt tile in LDS.
// ---------------------------------------------------------------------------
__global__ __launch_bounds__(256) void pv_kernel(
    const __bf16* __restrict__ P16, const __bf16* __restrict__ Vt16,
    float* __restrict__ out) {
  __shared__ __bf16 ldsB[2][64 * LDB];
  const int tid  = threadIdx.x;
  const int lane = tid & 31;
  const int wave = tid >> 5;
  const int half = lane >> 4;
  const int n    = lane & 15;
  const int mb   = half * 8;
  // blocks: b (8) x q-blk (8, 256 rows) x d-strip (16, 64 cols)
  int blk = blockIdx.x;
  int b   = blk / (8 * 16);
  int rem = blk % (8 * 16);
  int qb  = rem / 16;
  int dt  = rem % 16;
  int d0  = dt * 64;
  int gq  = qb * 256 + wave * 32;            // this wave's first q row
  const __bf16* Pb  = P16  + (size_t)b * SS * SS;
  const __bf16* Vtb = Vt16 + (size_t)b * HH * SS;

  v8f acc[2][4] = {};
  stage_tile(Vtb, SS, d0, 0, ldsB[0], tid);
  for (int it = 0; it < SS / 32; ++it) {
    int kk = it * 32;
    wait_async0();
    __syncthreads();
    if (it + 1 < SS / 32)
      stage_tile(Vtb, SS, d0, kk + 32, ldsB[(it + 1) & 1], tid);
    const __bf16* Bt = ldsB[it & 1];
    v16bf a0 = ldfrag(Pb, SS, gq + n, kk, half);
    v16bf a1 = ldfrag(Pb, SS, gq + 16 + n, kk, half);
#pragma unroll
    for (int j = 0; j < 4; ++j) {
      v16bf bf = ldfrag_lds(Bt, 16 * j + n, half);
      acc[0][j] = wmma_bf16(a0, bf, acc[0][j]);
      acc[1][j] = wmma_bf16(a1, bf, acc[1][j]);
    }
  }
#pragma unroll
  for (int i = 0; i < 2; ++i)
#pragma unroll
    for (int r = 0; r < 8; ++r) {
      size_t base = ((size_t)b * SS + gq + 16 * i + mb + r) * HH + d0 + n;
#pragma unroll
      for (int j = 0; j < 4; ++j) out[base + 16 * j] = acc[i][j][r];
    }
}

// ---------------------------------------------------------------------------
extern "C" void kernel_launch(void* const* d_in, const int* in_sizes, int n_in,
                              void* d_out, int out_size, void* d_ws, size_t ws_size,
                              hipStream_t stream) {
  const float* X  = (const float*)d_in[0];
  const unsigned char* mask = (const unsigned char*)d_in[1];  // jax bool: 1 byte
  const float* Wq = (const float*)d_in[2];
  const float* Wk = (const float*)d_in[3];
  const float* Wv = (const float*)d_in[4];
  float* out = (float*)d_out;

  char* ws = (char*)d_ws;
  __bf16* X16  = (__bf16*)(ws + 0);
  __bf16* W16  = (__bf16*)(ws + 33554432);
  __bf16* Q16  = (__bf16*)(ws + 39845888);
  __bf16* K16  = (__bf16*)(ws + 73400320);
  __bf16* Vt16 = (__bf16*)(ws + 106954752);
  float*  Mtile = (float*)(ws + 140509184);
  float*  Ltile = (float*)(ws + 142606336);
  __bf16* P16  = (__bf16*)(ws + 144703488);
  // total workspace required: 211,812,352 bytes

  {  // 1) convert X, Wq, Wk, Wv to bf16
    size_t tot4 = (size_t)BB * SS * HH / 4 + 3 * (size_t)HH * HH / 4;
    int blocks = (int)((tot4 + 255) / 256);
    cvt_kernel<<<blocks, 256, 0, stream>>>(X, Wq, Wk, Wv, X16, W16);
  }
  // 2) QKV projection: 3 x 64 x 16 = 3072 blocks (8 waves each)
  qkv_kernel<<<3072, 256, 0, stream>>>(X16, W16, Q16, K16, Vt16);
  // 3) scores + tile softmax: 8 x 64 x 32 = 16384 wave-jobs, 8 waves/block
  score_kernel<<<16384 / 8, 256, 0, stream>>>(Q16, K16, mask, P16, Mtile, Ltile);
  // 4) per-row reduction + P rescale
  finalize_kernel<<<BB * SS, 256, 0, stream>>>(Mtile, Ltile, P16);
  // 5) O = P @ V: 8 x 8 x 16 = 1024 blocks (8 waves each)
  pv_kernel<<<1024, 256, 0, stream>>>(P16, Vt16, out);
}
